// ScaledDotProductAttention_53214644797493
// MI455X (gfx1250) — matrix operand order repaired and checked by reading, not compiled, not run on hardware
//
#include <hip/hip_runtime.h>

typedef __attribute__((ext_vector_type(16))) _Float16 v16h;
typedef __attribute__((ext_vector_type(8)))  _Float16 v8h;
typedef __attribute__((ext_vector_type(4)))  _Float16 v4h;
typedef __attribute__((ext_vector_type(2)))  _Float16 v2h;
typedef __attribute__((ext_vector_type(8)))  float    v8f;
typedef __attribute__((ext_vector_type(4)))  float    f32x4;

// Problem constants (match reference)
constexpr int BB = 4, HH = 16, SS = 2048, DD = 64;
constexpr int KB = 64;              // keys per iteration
constexpr int WAVES = 4;            // waves per block
constexpr int ROWS_PER_BLOCK = 64;  // 4 waves x 16 query rows

__device__ __forceinline__ v8h cvt8(const float* __restrict__ p, float s) {
  f32x4 x0 = *(const f32x4*)p;
  f32x4 x1 = *(const f32x4*)(p + 4);
  v8h r;
#pragma unroll
  for (int i = 0; i < 4; ++i) r[i] = (_Float16)(x0[i] * s);
#pragma unroll
  for (int i = 0; i < 4; ++i) r[i + 4] = (_Float16)(x1[i] * s);
  return r;
}

__device__ __forceinline__ v8h ld8h(const _Float16* p) { return *(const v8h*)p; }

__device__ __forceinline__ v16h cat(v8h lo, v8h hi) {
  v16h r;
#pragma unroll
  for (int i = 0; i < 8; ++i) { r[i] = lo[i]; r[i + 8] = hi[i]; }
  return r;
}

__device__ __forceinline__ v8f wmma_f16(v16h a, v16h b, v8f c) {
  return __builtin_amdgcn_wmma_f32_16x16x32_f16(
      /*neg_a=*/false, a, /*neg_b=*/false, b,
      /*c_mod=*/(short)0, c, /*reuse_a=*/false, /*reuse_b=*/false);
}

__device__ __forceinline__ v2h hmax2(v2h a, v2h b) {
  v2h r;
  r[0] = (a[0] > b[0]) ? a[0] : b[0];
  r[1] = (a[1] > b[1]) ? a[1] : b[1];
  return r;
}

__global__ void __launch_bounds__(128, 1)
flash_attn_fwd(const float* __restrict__ Q, const float* __restrict__ K,
               const float* __restrict__ V, float* __restrict__ O) {
  // LDS: K tile row-major f16 (64x64), V tile transposed + key-permuted f16
  // (64 d x 64 kpos), per-wave P scratch (16 x 64 kpos).
  __shared__ __align__(16) _Float16 lds_k[KB * DD];          // 8 KB
  __shared__ __align__(16) _Float16 lds_vt[DD * KB];         // 8 KB
  __shared__ __align__(16) _Float16 lds_p[WAVES][16 * KB];   // 8 KB

  const int tid  = threadIdx.x;
  const int wave = tid >> 5;
  const int lane = tid & 31;
  const int col  = lane & 15;   // column / row position in wave32 layouts
  const int half = lane >> 4;

  const int bh    = blockIdx.y;
  const int qbase = blockIdx.x * ROWS_PER_BLOCK + wave * 16;
  const size_t base = (size_t)bh * SS * DD;

  // ---- Q tile (16x64) in A-layout; fold log2(e)/sqrt(D) so softmax is exp2 ----
  const float scale = 0.125f * 1.44269504088896340736f;
  const float* qrow = Q + base + (size_t)(qbase + col) * DD;
  v16h a0 = cat(cvt8(qrow +      half * 8, scale), cvt8(qrow + 16 + half * 8, scale));
  v16h a1 = cat(cvt8(qrow + 32 + half * 8, scale), cvt8(qrow + 48 + half * 8, scale));

  // Ones B-matrix: row-sum(P) via WMMA (replaces shuffle reductions for l).
  v16h ones;
#pragma unroll
  for (int i = 0; i < 16; ++i) ones[i] = (_Float16)1.0f;

  float mi[8], li[8];
#pragma unroll
  for (int v = 0; v < 8; ++v) { mi[v] = -1e30f; li[v] = 0.0f; }
  v8f oacc[4] = {v8f{}, v8f{}, v8f{}, v8f{}};

  // K staging: thread -> (key row 0..63, 32-wide d chunk 0..1)
  const int sr = tid >> 1;
  const int sc = (tid & 1) * 32;
  // V staging: thread -> (c 0..15, 8-wide d chunk 0..7); owns keys
  // {c, c+16, c+32, c+48} whose pi-positions 4c..4c+3 are contiguous -> b64 store.
  // pi(t*16+c) = 4c+t applied to P columns AND V key rows (P@V invariant).
  const int vc = tid & 15;
  const int vd = (tid >> 4) * 8;

  const int NITER = SS / KB;
  for (int kb = 0; kb < NITER; ++kb) {
    const int jbase = kb * KB;
    __syncthreads();  // previous iteration's readers done before overwrite

    // ---- Stage K tile: row-major f16 ----
    {
      const float* kp = K + base + (size_t)(jbase + sr) * DD + sc;
      *(v8h*)&lds_k[sr * DD + sc]      = cvt8(kp,      1.0f);
      *(v8h*)&lds_k[sr * DD + sc + 8]  = cvt8(kp + 8,  1.0f);
      *(v8h*)&lds_k[sr * DD + sc + 16] = cvt8(kp + 16, 1.0f);
      *(v8h*)&lds_k[sr * DD + sc + 24] = cvt8(kp + 24, 1.0f);
    }
    // ---- Stage V tile transposed, 4 keys packed per b64 store ----
    {
      const float* vrow = V + base + (size_t)jbase * DD + vd;
      f32x4 t0a = ((const f32x4*)(vrow + (vc +  0) * DD))[0];
      f32x4 t0b = ((const f32x4*)(vrow + (vc +  0) * DD))[1];
      f32x4 t1a = ((const f32x4*)(vrow + (vc + 16) * DD))[0];
      f32x4 t1b = ((const f32x4*)(vrow + (vc + 16) * DD))[1];
      f32x4 t2a = ((const f32x4*)(vrow + (vc + 32) * DD))[0];
      f32x4 t2b = ((const f32x4*)(vrow + (vc + 32) * DD))[1];
      f32x4 t3a = ((const f32x4*)(vrow + (vc + 48) * DD))[0];
      f32x4 t3b = ((const f32x4*)(vrow + (vc + 48) * DD))[1];
#pragma unroll
      for (int i = 0; i < 4; ++i) {
        v4h pa, pb;
        pa[0] = (_Float16)t0a[i]; pa[1] = (_Float16)t1a[i];
        pa[2] = (_Float16)t2a[i]; pa[3] = (_Float16)t3a[i];
        pb[0] = (_Float16)t0b[i]; pb[1] = (_Float16)t1b[i];
        pb[2] = (_Float16)t2b[i]; pb[3] = (_Float16)t3b[i];
        *(v4h*)&lds_vt[(vd + i)     * KB + 4 * vc] = pa;
        *(v4h*)&lds_vt[(vd + i + 4) * KB + 4 * vc] = pb;
      }
    }
    if (kb + 1 < NITER) {
      __builtin_prefetch(K + base + (size_t)(jbase + KB + sr) * DD + sc, 0, 0);
      __builtin_prefetch(V + base + (size_t)(jbase + KB + sr) * DD + sc, 0, 0);
    }
    __syncthreads();

    // ---- S = (Q*scale) @ K^T : four 16x16 score tiles ----
    v8f z{};
    v8f s[4];
#pragma unroll
    for (int t = 0; t < 4; ++t) {
      v16h bt0 = cat(ld8h(&lds_k[(t * 16 + col) * DD +      half * 8]),
                     ld8h(&lds_k[(t * 16 + col) * DD + 16 + half * 8]));
      v16h bt1 = cat(ld8h(&lds_k[(t * 16 + col) * DD + 32 + half * 8]),
                     ld8h(&lds_k[(t * 16 + col) * DD + 48 + half * 8]));
      s[t] = wmma_f16(a0, bt0, z);
      s[t] = wmma_f16(a1, bt1, s[t]);
    }

    // ---- Row max: local max3 then packed-f16 shuffle reduction ----
    // (softmax shift cancels numerator/denominator, so f16 max is exact enough;
    //  it only has to be uniform per row, which the packed reduce preserves)
    float lmax[8];
#pragma unroll
    for (int v = 0; v < 8; ++v)
      lmax[v] = fmaxf(fmaxf(s[0][v], s[1][v]), fmaxf(s[2][v], s[3][v]));
    v2h pk[4];
#pragma unroll
    for (int j = 0; j < 4; ++j) {
      pk[j][0] = (_Float16)lmax[2 * j];
      pk[j][1] = (_Float16)lmax[2 * j + 1];
    }
#pragma unroll
    for (int mask = 1; mask <= 8; mask <<= 1) {
#pragma unroll
      for (int j = 0; j < 4; ++j) {
        int w = __shfl_xor(__builtin_bit_cast(int, pk[j]), mask, 32);
        pk[j] = hmax2(pk[j], __builtin_bit_cast(v2h, w));
      }
    }

    // ---- Online softmax: exp2, packed b64 P stores, rescale O ----
    _Float16* pw = lds_p[wave];
    float corr[8];
#pragma unroll
    for (int v = 0; v < 8; ++v) {
      float mn = fmaxf(mi[v], (float)pk[v >> 1][v & 1]);
      corr[v]  = __builtin_amdgcn_exp2f(mi[v] - mn);
      mi[v]    = mn;
      v4h pp;
#pragma unroll
      for (int t = 0; t < 4; ++t)
        pp[t] = (_Float16)__builtin_amdgcn_exp2f(s[t][v] - mn);
      const int row = v + half * 8;
      *(v4h*)&pw[row * KB + 4 * col] = pp;  // pi-packed: one b64 store
      oacc[0][v] *= corr[v];
      oacc[1][v] *= corr[v];
      oacc[2][v] *= corr[v];
      oacc[3][v] *= corr[v];
    }

    // ---- P (C-layout) -> A-layout (permuted key positions) ----
    v16h ap0 = cat(ld8h(&pw[col * KB +      half * 8]),
                   ld8h(&pw[col * KB + 16 + half * 8]));
    v16h ap1 = cat(ld8h(&pw[col * KB + 32 + half * 8]),
                   ld8h(&pw[col * KB + 48 + half * 8]));

    // ---- Row sums of P via WMMA with ones ----
    v8f lsum = wmma_f16(ap0, ones, z);
    lsum     = wmma_f16(ap1, ones, lsum);
#pragma unroll
    for (int v = 0; v < 8; ++v) li[v] = li[v] * corr[v] + lsum[v];

    // ---- O += P @ V : 4 d-segments x 2 key-halves ----
#pragma unroll
    for (int dseg = 0; dseg < 4; ++dseg) {
      v16h vb0 = cat(ld8h(&lds_vt[(dseg * 16 + col) * KB +      half * 8]),
                     ld8h(&lds_vt[(dseg * 16 + col) * KB + 16 + half * 8]));
      oacc[dseg] = wmma_f16(ap0, vb0, oacc[dseg]);
      v16h vb1 = cat(ld8h(&lds_vt[(dseg * 16 + col) * KB + 32 + half * 8]),
                     ld8h(&lds_vt[(dseg * 16 + col) * KB + 48 + half * 8]));
      oacc[dseg] = wmma_f16(ap1, vb1, oacc[dseg]);
    }
  }

  // ---- Epilogue: O * (1/l), C-layout scatter (coalesced across lanes) ----
  float inv[8];
#pragma unroll
  for (int v = 0; v < 8; ++v) inv[v] = 1.0f / li[v];
#pragma unroll
  for (int dseg = 0; dseg < 4; ++dseg) {
#pragma unroll
    for (int v = 0; v < 8; ++v) {
      const int row = v + half * 8;
      O[base + (size_t)(qbase + row) * DD + dseg * 16 + col] = oacc[dseg][v] * inv[v];
    }
  }
}

extern "C" void kernel_launch(void* const* d_in, const int* in_sizes, int n_in,
                              void* d_out, int out_size, void* d_ws, size_t ws_size,
                              hipStream_t stream) {
  const float* Q = (const float*)d_in[0];
  const float* K = (const float*)d_in[1];
  const float* V = (const float*)d_in[2];
  float* O = (float*)d_out;
  (void)in_sizes; (void)n_in; (void)out_size; (void)d_ws; (void)ws_size;

  dim3 grid(SS / ROWS_PER_BLOCK, BB * HH);  // (32, 64)
  dim3 block(WAVES * 32);                   // 128 threads = 4 waves (wave32)
  flash_attn_fwd<<<grid, block, 0, stream>>>(Q, K, V, O);
}
